// DenseRelativeLoc_65154653880342
// MI455X (gfx1250) — matrix operand order repaired and checked by reading, not compiled, not run on hardware
//
#include <hip/hip_runtime.h>
#include <hip/hip_bf16.h>

typedef __attribute__((ext_vector_type(16))) _Float16 v16h;
typedef __attribute__((ext_vector_type(8)))  _Float16 v8h;
typedef __attribute__((ext_vector_type(8)))  float    v8f;

#define B_   64
#define C_   768
#define H_   56
#define S_   256
#define M_   (B_ * S_)      // 16384 rows
#define K1_  (2 * C_)       // 1536
#define N1_  256
#define PLANE_ (H_ * H_)    // 3136
#define KC_  128            // K-chunk staged in LDS
#define KSTEPS_ (KC_ / 32)  // 4 WMMA k-steps per chunk
#define LDSP_ (KC_ + 8)     // padded LDS row stride (halves): 272B -> conflict-free

// ---------------------------------------------------------------------------
// Transpose + f32->f16 convert: W [K,N] row-major  ->  Wt [N,K] f16 row-major
// ---------------------------------------------------------------------------
__global__ __launch_bounds__(256) void wt_convert_kernel(
    const float* __restrict__ W, _Float16* __restrict__ Wt, int K, int N) {
  int id = blockIdx.x * 256 + threadIdx.x;
  if (id >= K * N) return;
  int k = id / N;
  int n = id - k * N;
  Wt[(size_t)n * K + k] = (_Float16)W[id];
}

// ---------------------------------------------------------------------------
// Gather: one block per (b,s). Threads stride channels; writes are coalesced
// f16 halves; reads are scattered 4B but share channel planes across the 256
// concurrent blocks of the same batch (L2-resident 12.5KB planes).
// Also emits deltaxy directly.
// ---------------------------------------------------------------------------
__global__ __launch_bounds__(256) void gather_kernel(
    const float* __restrict__ x, const int* __restrict__ pxs,
    const int* __restrict__ pys, _Float16* __restrict__ feats,
    float* __restrict__ delta_out) {
  const int bs = blockIdx.x;          // 0..16383
  const int b  = bs >> 8;
  const int t  = threadIdx.x;         // 0..255

  const int px0 = pxs[bs * 2 + 0], px1 = pxs[bs * 2 + 1];
  const int py0 = pys[bs * 2 + 0], py1 = pys[bs * 2 + 1];
  const int idxx = px0 * H_ + px1;
  const int idxy = py0 * H_ + py1;

  const float* xb = x + (size_t)b * C_ * PLANE_;
  _Float16* frow = feats + (size_t)bs * K1_;

#pragma unroll
  for (int k = 0; k < 3; ++k) {
    const int c = t + k * 256;
    const size_t off = (size_t)c * PLANE_;
    frow[c]        = (_Float16)xb[off + idxx];
    frow[C_ + c]   = (_Float16)xb[off + idxy];
  }

  if (t < 2) {
    delta_out[bs * 2 + t] =
        (float)(pxs[bs * 2 + t] - pys[bs * 2 + t]) + (float)(H_ - 1);
  }
}

// ---------------------------------------------------------------------------
// A fragment loader: 16x32 f16 tile, per documented wave32 layout.
//   lane L = row M=L%16; halves 0..7 = K(k0 + 8*(L/16)..),
//   halves 8..15 = K(k0+16 + 8*(L/16)..)   -> two 16B loads
// ---------------------------------------------------------------------------
__device__ __forceinline__ v16h load_afrag(const _Float16* p, int half) {
  const v8h lo = *(const v8h*)(p + half * 8);
  const v8h hi = *(const v8h*)(p + 16 + half * 8);
  v16h a;
#pragma unroll
  for (int i = 0; i < 8; ++i) { a[i] = lo[i]; a[i + 8] = hi[i]; }
  return a;
}

// ---------------------------------------------------------------------------
// WMMA GEMM with LDS-staged B: C[M,N] = relu?(A[M,K] * Bt[N,K]^T + bias), f16 out.
// Block = 8 waves = 128(M) x 64(N); waves share one 64-col B panel staged in
// LDS per 128-K chunk (one cooperative load -> 8x less global B traffic).
// Wave computes a 16x64 tile: per 32-K step, all 4 B fragments are batch-
// loaded from LDS, then 4 independent v_wmma_f32_16x16x32_f16 run back-to-
// back; the A fragment of the next step is preloaded during the current one.
// Grids divide exactly -> no predication around WMMA (EXEC all-1s).
//
//   B 32x16 f16 : lane L = col N=L%16; halves 0..7 -> K = k0 + 8*(L/16) + e, halves 8..15 -> K = k0+16 + 8*(L/16) + (e-8)
//   C/D 16x16 f32: VGPR r, lane L -> M = r + 8*(L/16), N = L%16
// ---------------------------------------------------------------------------
__global__ __launch_bounds__(256) void gemm_relu_f16_kernel(
    const _Float16* __restrict__ A, const _Float16* __restrict__ Bt,
    const float* __restrict__ bias, _Float16* __restrict__ Cout,
    int M, int N, int K, int do_relu) {
  __shared__ _Float16 Bs[64][LDSP_];            // 17.4 KB

  const int tid  = threadIdx.x;
  const int lane = tid & 31;
  const int wave = tid >> 5;                    // 0..7
  const int half = lane >> 4;                   // 0 or 1
  const int l16  = lane & 15;

  const int m0 = blockIdx.x * 128 + wave * 16;  // 16 rows per wave
  const int n0 = blockIdx.y * 64;               // 64 cols, shared by block

  const _Float16* arow = A + (size_t)(m0 + l16) * K;

  // cooperative B-chunk loader mapping: 64 rows x 128 halves, 32 halves/thread
  const int brow = tid >> 2;                    // 0..63
  const int bseg = (tid & 3) * 32;              // halves offset in chunk

  v8f acc[4];
#pragma unroll
  for (int j = 0; j < 4; ++j) acc[j] = (v8f){};

  for (int kc = 0; kc < K; kc += KC_) {
    // ---- stage B chunk into LDS ----
    const _Float16* bsrc = Bt + (size_t)(n0 + brow) * K + kc + bseg;
    *(uint4*)(&Bs[brow][bseg + 0])  = *(const uint4*)(bsrc + 0);
    *(uint4*)(&Bs[brow][bseg + 8])  = *(const uint4*)(bsrc + 8);
    *(uint4*)(&Bs[brow][bseg + 16]) = *(const uint4*)(bsrc + 16);
    *(uint4*)(&Bs[brow][bseg + 24]) = *(const uint4*)(bsrc + 24);
    __syncthreads();

    // prefetch next A chunk into near caches
    if (kc + KC_ < K) __builtin_prefetch(arow + kc + KC_, 0, 3);

    // ---- 4 WMMA k-steps, A fragment pipelined one step ahead ----
    v16h a_cur = load_afrag(arow + kc, half);
#pragma unroll
    for (int step = 0; step < KSTEPS_; ++step) {
      const int kk = step * 32;
      const v16h a_nxt = (step + 1 < KSTEPS_)
                             ? load_afrag(arow + kc + kk + 32, half)
                             : a_cur;

      v16h bfrag[4];
#pragma unroll
      for (int j = 0; j < 4; ++j) {
        const _Float16* bp = &Bs[j * 16 + l16][kk + half * 8];
        const v8h lo = *(const v8h*)(bp);
        const v8h hi = *(const v8h*)(bp + 16);
#pragma unroll
        for (int i = 0; i < 8; ++i) { bfrag[j][i] = lo[i]; bfrag[j][i + 8] = hi[i]; }
      }

#pragma unroll
      for (int j = 0; j < 4; ++j) {
        acc[j] = __builtin_amdgcn_wmma_f32_16x16x32_f16(
            false, a_cur, false, bfrag[j], (short)0, acc[j], false, false);
      }
      a_cur = a_nxt;
    }
    __syncthreads();
  }

#pragma unroll
  for (int j = 0; j < 4; ++j) {
    const int n = n0 + j * 16 + l16;
    const float bn = bias[n];
#pragma unroll
    for (int r = 0; r < 8; ++r) {
      const int m = m0 + r + 8 * half;
      float v = acc[j][r] + bn;
      if (do_relu) v = v > 0.0f ? v : 0.0f;
      Cout[(size_t)m * N + n] = (_Float16)v;
    }
  }
}

// ---------------------------------------------------------------------------
// Head: pred[row,j] = sum_k hdn2[row,k] * W3[k,j] + b3[j]   (N=2, plain FMA)
// ---------------------------------------------------------------------------
__global__ __launch_bounds__(256) void head_kernel(
    const _Float16* __restrict__ hdn2, const float* __restrict__ W3,
    const float* __restrict__ b3, float* __restrict__ pred) {
  const int id  = blockIdx.x * 256 + threadIdx.x;  // 0..32767
  const int row = id >> 1;
  const int j   = id & 1;
  const _Float16* hr = hdn2 + (size_t)row * N1_;
  float acc = b3[j];
#pragma unroll 8
  for (int k = 0; k < N1_; ++k) acc += (float)hr[k] * W3[k * 2 + j];
  pred[id] = acc;
}

// ---------------------------------------------------------------------------
// Launch
// ---------------------------------------------------------------------------
extern "C" void kernel_launch(void* const* d_in, const int* in_sizes, int n_in,
                              void* d_out, int out_size, void* d_ws, size_t ws_size,
                              hipStream_t stream) {
  const float* x   = (const float*)d_in[0];
  const int*   pxs = (const int*)d_in[1];
  const int*   pys = (const int*)d_in[2];
  const float* W1  = (const float*)d_in[3];
  const float* b1  = (const float*)d_in[4];
  const float* W2  = (const float*)d_in[5];
  const float* b2  = (const float*)d_in[6];
  const float* W3  = (const float*)d_in[7];
  const float* b3  = (const float*)d_in[8];

  float* pred  = (float*)d_out;                 // [16384, 2]
  float* delta = (float*)d_out + (size_t)M_ * 2;

  // Workspace layout
  char* ws = (char*)d_ws;
  _Float16* feats = (_Float16*)(ws);                               // 48 MB
  _Float16* hdn1  = (_Float16*)(ws + (size_t)M_ * K1_ * 2);        //  8 MB
  _Float16* hdn2  = (_Float16*)(ws + (size_t)M_ * K1_ * 2
                                   + (size_t)M_ * N1_ * 2);        //  8 MB
  _Float16* Wt1   = (_Float16*)(ws + (size_t)M_ * K1_ * 2
                                   + (size_t)M_ * N1_ * 2 * 2);    // 0.75 MB
  _Float16* Wt2   = (_Float16*)((char*)Wt1 + (size_t)K1_ * N1_ * 2);

  // 1) weight convert + transpose to [N,K] f16
  wt_convert_kernel<<<(K1_ * N1_ + 255) / 256, 256, 0, stream>>>(W1, Wt1, K1_, N1_);
  wt_convert_kernel<<<(N1_ * N1_ + 255) / 256, 256, 0, stream>>>(W2, Wt2, N1_, N1_);

  // 2) gather + deltaxy
  gather_kernel<<<M_, 256, 0, stream>>>(x, pxs, pys, feats, delta);

  // 3) layer 1: [16384,1536] @ [1536,256] + b1, relu
  gemm_relu_f16_kernel<<<dim3(M_ / 128, N1_ / 64), 256, 0, stream>>>(
      feats, Wt1, b1, hdn1, M_, N1_, K1_, 1);

  // 4) layer 2: [16384,256] @ [256,256] + b2, relu
  gemm_relu_f16_kernel<<<dim3(M_ / 128, N1_ / 64), 256, 0, stream>>>(
      hdn1, Wt2, b2, hdn2, M_, N1_, N1_, 1);

  // 5) head: [16384,256] @ [256,2] + b3
  head_kernel<<<(M_ * 2) / 256, 256, 0, stream>>>(hdn2, W3, b3, pred);
}
